// PWCFlow_26560077758497
// MI455X (gfx1250) — matrix operand order repaired and checked by reading, not compiled, not run on hardware
//
#include <hip/hip_runtime.h>
#include <stdint.h>

typedef __attribute__((ext_vector_type(16))) __bf16 v16bf_t;
typedef __attribute__((ext_vector_type(8)))  float  v8f_t;

#define ALPHA 0.1f

__device__ inline unsigned short f2bf(float f) {
  union { float f; unsigned u; } x; x.f = f;
  unsigned r = x.u + 0x7FFFu + ((x.u >> 16) & 1u);   // round-to-nearest-even
  return (unsigned short)(r >> 16);
}
__device__ inline float bf2f(unsigned short u) {
  union { unsigned u; float f; } x; x.u = ((unsigned)u) << 16; return x.f;
}

union Frag { v16bf_t v; uint4 q[2]; };

// ---------------------------------------------------------------------------
// Implicit-GEMM 3x3 conv (SAME, dilation d) with bf16 WMMA, fp32 accumulate.
// One wave per 32-pixel x 16-cout tile (two 16x16 A tiles sharing B fragments,
// two independent accumulator chains for XDL overlap).
// Tap-major loop: per-tap the OOB predicate selects either the real pixel base
// or a zero page (Z), so the inner K-loop is branchless: 6 unconditional b128
// loads + 2 WMMAs per chunk, no EXEC masking, no register zero-fill.
// ---------------------------------------------------------------------------
struct ConvParams {
  const unsigned short* __restrict__ X; long Cx;  // input (channel-fastest)
  const unsigned short* __restrict__ Z;           // zero page (>= 1024 u16)
  int cin;                                        // padded GEMM cin (mult 32)
  int dil;
  const unsigned short* __restrict__ Wp;          // packed weights
  const float* __restrict__ bias; int cout_real;
  int B, H, W;
  unsigned short* __restrict__ Ybf; long Ycx;     // optional bf16 out (pre-offset)
  float* __restrict__ Yf; long Yst;               // optional fp32 out
  const float* __restrict__ Add;                  // optional residual (stride 2)
  int relu;
};

__global__ void __launch_bounds__(32) conv3x3_wmma(ConvParams p) {
  const int lane = threadIdx.x;
  const int hi8  = (lane >> 4) & 1;
  const long npix = (long)p.B * p.H * p.W;
  const long pixBase = (long)blockIdx.x * 32;
  const int nTile = blockIdx.y;
  const int hw = p.H * p.W;

  long ap0 = pixBase + (lane & 15);
  long ap1 = ap0 + 16;
  const bool pv0 = ap0 < npix, pv1 = ap1 < npix;
  long q0 = pv0 ? ap0 : 0, q1 = pv1 ? ap1 : 0;
  const int b0 = (int)(q0 / hw), r0 = (int)(q0 % hw);
  const int y0 = r0 / p.W, x0 = r0 % p.W;
  const int b1 = (int)(q1 / hw), r1 = (int)(q1 % hw);
  const int y1 = r1 / p.W, x1 = r1 % p.W;

  v8f_t acc0 = {0.f,0.f,0.f,0.f,0.f,0.f,0.f,0.f};
  v8f_t acc1 = {0.f,0.f,0.f,0.f,0.f,0.f,0.f,0.f};
  const int cpt = p.cin >> 5;                       // K-chunks per tap
  const unsigned short* wptr =
      p.Wp + ((long)nTile * (9 * cpt) * 32 + lane) * 16;
  const unsigned short* zp = p.Z + hi8 * 16;

#pragma unroll
  for (int tap = 0; tap < 9; ++tap) {
    const int kh = tap / 3, kw = tap % 3;           // literals after unroll
    const int yy0 = y0 + (kh - 1) * p.dil, xx0 = x0 + (kw - 1) * p.dil;
    const int yy1 = y1 + (kh - 1) * p.dil, xx1 = x1 + (kw - 1) * p.dil;
    const bool v0 = pv0 && yy0 >= 0 && yy0 < p.H && xx0 >= 0 && xx0 < p.W;
    const bool v1 = pv1 && yy1 >= 0 && yy1 < p.H && xx1 >= 0 && xx1 < p.W;
    const unsigned short* s0 = v0 ?
        p.X + (((long)b0 * p.H + yy0) * p.W + xx0) * p.Cx + hi8 * 16 : zp;
    const unsigned short* s1 = v1 ?
        p.X + (((long)b1 * p.H + yy1) * p.W + xx1) * p.Cx + hi8 * 16 : zp;
    for (int cc = 0; cc < cpt; ++cc) {
      Frag w, a0, a1;
      w.q[0] = *(const uint4*)(wptr);
      w.q[1] = *(const uint4*)(wptr + 8);
      __builtin_prefetch(wptr + 512, 0, 1);         // -> global_prefetch_b8
      wptr += 512;
      a0.q[0] = *(const uint4*)(s0);
      a0.q[1] = *(const uint4*)(s0 + 8);
      a1.q[0] = *(const uint4*)(s1);
      a1.q[1] = *(const uint4*)(s1 + 8);
      s0 += 32; s1 += 32;
      acc0 = __builtin_amdgcn_wmma_f32_16x16x32_bf16(false, a0.v, false, w.v,
                                                     (short)0, acc0, false, false);
      acc1 = __builtin_amdgcn_wmma_f32_16x16x32_bf16(false, a1.v, false, w.v,
                                                     (short)0, acc1, false, false);
    }
  }

  const int n = nTile * 16 + (lane & 15);
  if (n >= p.cout_real) return;
  const float bv = p.bias ? p.bias[n] : 0.f;
#pragma unroll
  for (int t = 0; t < 2; ++t) {
    const v8f_t& a = t ? acc1 : acc0;
#pragma unroll
    for (int r = 0; r < 8; ++r) {
      long pm = pixBase + t * 16 + r + hi8 * 8;     // D layout: m = r + 8*(L>=16)
      if (pm >= npix) continue;
      float v = a[r] + bv;
      if (p.Add)  v += p.Add[pm * 2 + n];
      if (p.relu) v = fmaxf(v, ALPHA * v);
      if (p.Yf)   p.Yf[pm * p.Yst + n] = v;
      if (p.Ybf)  p.Ybf[pm * p.Ycx + n] = f2bf(v);
    }
  }
}

// ---------------------------------------------------------------------------
// Weight packing: HWIO fp32 -> bf16 fragments [ntile][chunk][lane][16],
// mapping padded GEMM channels -> reference concat channels via segments.
// ---------------------------------------------------------------------------
struct PackDesc {
  int cin, cin_real, cout_real, nseg;
  int moff[10], roff[10], slen[10];
};
__global__ void pack_weights(const float* w, unsigned short* out, PackDesc d,
                             int ntiles, int chunks) {
  long idx = (long)blockIdx.x * blockDim.x + threadIdx.x;
  long total = (long)ntiles * chunks * 512;
  if (idx >= total) return;
  int hi = idx & 15;
  int lane = (int)((idx >> 4) & 31);
  long t = idx >> 9;
  int chunk = (int)(t % chunks);
  int ntile = (int)(t / chunks);
  int k = chunk * 32 + ((lane >> 4) & 1) * 16 + hi;
  int n = ntile * 16 + (lane & 15);
  int tap = k / d.cin, c = k % d.cin;
  float v = 0.f;
  if (n < d.cout_real) {
    int r = -1;
    for (int s = 0; s < d.nseg; ++s) {
      int o = d.moff[s];
      if (c >= o && c < o + d.slen[s]) { r = d.roff[s] + (c - o); break; }
    }
    if (r >= 0) v = w[((long)tap * d.cin_real + r) * d.cout_real + n];
  }
  out[idx] = f2bf(v);
}

// ---------------------------------------------------------------------------
// Small scalar kernels
// ---------------------------------------------------------------------------
__global__ void zero_u16x8(unsigned short* p, long n8) {   // n multiple of 8
  long i = (long)blockIdx.x * blockDim.x + threadIdx.x;
  if (i < n8) ((uint4*)p)[i] = uint4{0,0,0,0};
}
__global__ void zero_f32(float* p, long n) {
  long i = (long)blockIdx.x * blockDim.x + threadIdx.x;
  if (i < n) p[i] = 0.f;
}
__global__ void add_inplace(float* dst, const float* a, long n) {
  long i = (long)blockIdx.x * blockDim.x + threadIdx.x;
  if (i < n) dst[i] += a[i];
}
__global__ void f32_to_x(const float* src, unsigned short* dst, long dStride,
                         int dOff, long npix, int C) {
  long i = (long)blockIdx.x * blockDim.x + threadIdx.x;
  if (i >= npix * C) return;
  long pix = i / C; int c = (int)(i % C);
  dst[pix * dStride + dOff + c] = f2bf(src[i]);
}
__global__ void copy_bf_x(const unsigned short* src, long sStride, int sOff,
                          unsigned short* dst, long dStride, int dOff,
                          long npix, int C) {
  long i = (long)blockIdx.x * blockDim.x + threadIdx.x;
  if (i >= npix * C) return;
  long pix = i / C; int c = (int)(i % C);
  dst[pix * dStride + dOff + c] = src[pix * sStride + sOff + c];
}

// per-image sum & sumsq reduction (LDS + global atomics)
__global__ void reduce_sums(const float* t, long perImage, float* acc2) {
  __shared__ float ss[128], sq[128];
  int b = blockIdx.y;
  const float* p = t + (long)b * perImage;
  float s = 0.f, q = 0.f;
  for (long i = (long)blockIdx.x * blockDim.x + threadIdx.x; i < perImage;
       i += (long)gridDim.x * blockDim.x) {
    float v = p[i]; s += v; q += v * v;
  }
  ss[threadIdx.x] = s; sq[threadIdx.x] = q; __syncthreads();
  for (int o = 64; o > 0; o >>= 1) {
    if ((int)threadIdx.x < o) { ss[threadIdx.x] += ss[threadIdx.x + o]; sq[threadIdx.x] += sq[threadIdx.x + o]; }
    __syncthreads();
  }
  if (threadIdx.x == 0) { atomicAdd(&acc2[b * 2], ss[0]); atomicAdd(&acc2[b * 2 + 1], sq[0]); }
}
__global__ void finalize_stats(const float* a1, const float* a2, long perImage,
                               int B, float* stats) {
  if (threadIdx.x | blockIdx.x) return;
  float m1 = 0, m2 = 0, v1 = 0, v2 = 0;
  for (int b = 0; b < B; ++b) {
    float u1 = a1[b*2] / perImage, q1 = a1[b*2+1] / perImage;
    float u2 = a2[b*2] / perImage, q2 = a2[b*2+1] / perImage;
    m1 += u1; m2 += u2; v1 += q1 - u1*u1; v2 += q2 - u2*u2;
  }
  float m = 0.5f * (m1 + m2) / B;
  float v = 0.5f * (v1 + v2) / B;
  stats[0] = m; stats[1] = rsqrtf(v + 1e-16f);
}

// cost volume: normalized correlation, 81 shifts, leaky-relu, bf16 -> X@cvOff
__global__ void cost_volume(const float* f1, const float* w2, const float* stats,
                            unsigned short* X, long Cx, int cvOff,
                            int B, int H, int W, int C) {
  long idx = (long)blockIdx.x * blockDim.x + threadIdx.x;
  long total = (long)B * H * W * 81;
  if (idx >= total) return;
  int shift = (int)(idx % 81);
  long pix = idx / 81;
  int b = (int)(pix / (H * W)); int r = (int)(pix % (H * W));
  int y = r / W, x = r % W;
  int yy = y + shift / 9 - 4, xx = x + shift % 9 - 4;
  float m = stats[0], inv = stats[1];
  float s = 0.f;
  if (yy >= 0 && yy < H && xx >= 0 && xx < W) {
    const float* a = f1 + pix * C;
    const float* p = w2 + (((long)b * H + yy) * W + xx) * C;
    for (int c = 0; c < C; c += 4) {
      float4 av = *(const float4*)(a + c);
      float4 bv = *(const float4*)(p + c);
      s += (av.x - m) * (bv.x - m) + (av.y - m) * (bv.y - m) +
           (av.z - m) * (bv.z - m) + (av.w - m) * (bv.w - m);
    }
    s = s * inv * inv / C;
  }
  s = fmaxf(s, ALPHA * s);
  X[pix * Cx + cvOff + shift] = f2bf(s);
}

// bilinear warp with zero outside (per-corner validity), fp32 out
__global__ void warp_kernel(const float* f2, const float* flow, float* out,
                            int B, int H, int W, int C) {
  long pix = (long)blockIdx.x * blockDim.x + threadIdx.x;
  long np = (long)B * H * W;
  if (pix >= np) return;
  int b = (int)(pix / (H * W)); int r = (int)(pix % (H * W));
  int y = r / W, x = r % W;
  float ci = y + flow[pix * 2], cj = x + flow[pix * 2 + 1];
  float fi = floorf(ci), fj = floorf(cj);
  int i0 = (int)fi, j0 = (int)fj;
  float wi = ci - fi, wj = cj - fj;
  int i0c = min(max(i0, 0), H - 1), i1c = min(max(i0 + 1, 0), H - 1);
  int j0c = min(max(j0, 0), W - 1), j1c = min(max(j0 + 1, 0), W - 1);
  float m00 = (i0 >= 0 && i0 < H && j0 >= 0 && j0 < W) ? 1.f : 0.f;
  float m01 = (i0 >= 0 && i0 < H && j0+1 >= 0 && j0+1 < W) ? 1.f : 0.f;
  float m10 = (i0+1 >= 0 && i0+1 < H && j0 >= 0 && j0 < W) ? 1.f : 0.f;
  float m11 = (i0+1 >= 0 && i0+1 < H && j0+1 >= 0 && j0+1 < W) ? 1.f : 0.f;
  const float* base = f2 + (long)b * H * W * C;
  long o00 = ((long)i0c * W + j0c) * C, o01 = ((long)i0c * W + j1c) * C;
  long o10 = ((long)i1c * W + j0c) * C, o11 = ((long)i1c * W + j1c) * C;
  float w00 = (1-wi)*(1-wj)*m00, w01 = (1-wi)*wj*m01, w10 = wi*(1-wj)*m10, w11 = wi*wj*m11;
  for (int c = 0; c < C; ++c)
    out[pix * C + c] = w00*base[o00+c] + w01*base[o01+c] + w10*base[o10+c] + w11*base[o11+c];
}

// bilinear 2x upsample (half-pixel), x2.0 scale, 2 channels
__global__ void upsample2x_flow(const float* fin, float* fout, int B, int H, int W) {
  int H2 = 2 * H, W2 = 2 * W;
  long np = (long)B * H2 * W2;
  long pix = (long)blockIdx.x * blockDim.x + threadIdx.x;
  if (pix >= np) return;
  int b = (int)(pix / (H2 * W2)); int r = (int)(pix % (H2 * W2));
  int y2 = r / W2, x2 = r % W2;
  float sy = (y2 + 0.5f) * 0.5f - 0.5f, sx = (x2 + 0.5f) * 0.5f - 0.5f;
  float fy = floorf(sy), fx = floorf(sx);
  int i0 = (int)fy, j0 = (int)fx;
  float wy = sy - fy, wx = sx - fx;
  int i0c = min(max(i0, 0), H - 1), i1c = min(max(i0 + 1, 0), H - 1);
  int j0c = min(max(j0, 0), W - 1), j1c = min(max(j0 + 1, 0), W - 1);
  const float* base = fin + (long)b * H * W * 2;
  for (int c = 0; c < 2; ++c) {
    float v00 = base[((long)i0c * W + j0c) * 2 + c], v01 = base[((long)i0c * W + j1c) * 2 + c];
    float v10 = base[((long)i1c * W + j0c) * 2 + c], v11 = base[((long)i1c * W + j1c) * 2 + c];
    fout[pix * 2 + c] = 2.f * ((1-wy)*((1-wx)*v00 + wx*v01) + wy*((1-wx)*v10 + wx*v11));
  }
}

// 4x4 stride-2 transposed conv, 32->32 (context up); bf16 in, fp32 out
__global__ void deconv4x4(const unsigned short* X, long Cx, int coff,
                          const float* w, const float* bias, float* out,
                          int B, int H, int W) {
  int H2 = 2 * H, W2 = 2 * W;
  long np = (long)B * H2 * W2;
  long pix = (long)blockIdx.x * blockDim.x + threadIdx.x;
  if (pix >= np) return;
  int b = (int)(pix / (H2 * W2)); int r = (int)(pix % (H2 * W2));
  int y2 = r / W2, x2 = r % W2;
  float acc[32];
  for (int c = 0; c < 32; ++c) acc[c] = bias[c];
  for (int kh = 0; kh < 4; ++kh) {
    int ty = y2 + 1 - kh;
    if (ty < 0 || (ty & 1)) continue;
    int iy = ty >> 1; if (iy >= H) continue;
    for (int kw = 0; kw < 4; ++kw) {
      int tx = x2 + 1 - kw;
      if (tx < 0 || (tx & 1)) continue;
      int ix = tx >> 1; if (ix >= W) continue;
      const unsigned short* src = X + (((long)b * H + iy) * W + ix) * Cx + coff;
      const float* wp = w + ((kh * 4 + kw) * 32) * 32;
      for (int ci = 0; ci < 32; ++ci) {
        float v = bf2f(src[ci]);
        for (int co = 0; co < 32; ++co) acc[co] += v * wp[ci * 32 + co];
      }
    }
  }
  for (int c = 0; c < 32; ++c) out[pix * 32 + c] = acc[c];
}

// ---------------------------------------------------------------------------
// Host orchestration
// ---------------------------------------------------------------------------
namespace {

struct Bump {
  char* base; size_t off;
  void* alloc(size_t bytes) {
    off = (off + 255) & ~(size_t)255;
    void* r = base + off; off += bytes; return r;
  }
};

struct LvlCfg { int Cx, cvOff, f1Off, o[5]; };
// channel offsets, all 32-aligned; ctx@0, flow@32 for levels 1..3
static const LvlCfg LCFG[5] = {
  {0,0,0,{0,0,0,0,0}},
  {640, 64, 160, {192, 320, 448, 544, 608}},   // level 1 (f1: 32 ch)
  {672, 64, 160, {224, 352, 480, 576, 640}},   // level 2 (f1: 64 ch)
  {704, 64, 160, {256, 384, 512, 608, 672}},   // level 3 (f1: 96 ch)
  {672,  0,  96, {224, 352, 480, 576, 640}},   // level 4 (cv@0, f1: 128 ch)
};
static const int CHANS_L[5] = {16, 32, 64, 96, 128};
static const int COUT[5] = {128, 128, 96, 64, 32};
static const int CINREAL0[5] = {0, 147, 179, 211, 209};

// d_in index helpers (jax tree-flatten: alphabetical keys)
static inline int flow_b(int lvl, int k) { return 16 + (lvl - 1) * 12 + k; }
static inline int flow_w(int lvl, int k) { return 16 + (lvl - 1) * 12 + 6 + k; }
static inline int ctx_b(int lvl) { return 10 + (lvl - 2) * 2; }
static inline int ctx_w(int lvl) { return 11 + (lvl - 2) * 2; }
static inline int ref_b(int k) { return 64 + k; }
static inline int ref_w(int k) { return 71 + k; }

static unsigned short* do_pack(hipStream_t s, Bump& al, const float* w,
                               int cin, int cout_real, int cin_real,
                               int nseg, const int* moff, const int* roff,
                               const int* slen) {
  int ntiles = (cout_real + 15) / 16, chunks = (9 * cin) / 32;
  long n = (long)ntiles * chunks * 512;
  unsigned short* out = (unsigned short*)al.alloc((size_t)n * 2);
  PackDesc d{}; d.cin = cin; d.cin_real = cin_real; d.cout_real = cout_real; d.nseg = nseg;
  for (int i = 0; i < nseg; ++i) { d.moff[i] = moff[i]; d.roff[i] = roff[i]; d.slen[i] = slen[i]; }
  pack_weights<<<dim3((unsigned)((n + 255) / 256)), 256, 0, s>>>(w, out, d, ntiles, chunks);
  return out;
}

static void launch_conv(hipStream_t s, const unsigned short* X, long Cx,
                        const unsigned short* Z, int cin,
                        int dil, const unsigned short* Wp, const float* bias,
                        int cout_real, int B, int H, int W,
                        unsigned short* Ybf, long Ycx, float* Yf, long Yst,
                        const float* Add, int relu) {
  ConvParams p{X, Cx, Z, cin, dil, Wp, bias, cout_real, B, H, W, Ybf, Ycx, Yf, Yst, Add, relu};
  long np = (long)B * H * W;
  dim3 g((unsigned)((np + 31) / 32), (unsigned)((cout_real + 15) / 16));
  conv3x3_wmma<<<g, 32, 0, s>>>(p);
}

static inline dim3 g1(long n, int t) { return dim3((unsigned)((n + t - 1) / t)); }

} // namespace

extern "C" void kernel_launch(void* const* d_in, const int* in_sizes, int n_in,
                              void* d_out, int out_size, void* d_ws, size_t ws_size,
                              hipStream_t stream) {
  (void)in_sizes; (void)n_in; (void)out_size; (void)ws_size;
  const int B = 8;
  const int Hl[5] = {0, 96, 48, 24, 12};
  const int Wl[5] = {0, 128, 64, 32, 16};
  long P[5]; for (int l = 1; l <= 4; ++l) P[l] = (long)B * Hl[l] * Wl[l];
  const long outOff[5] = {0, 0, P[1] * 2, P[1] * 2 + P[2] * 2, P[1] * 2 + P[2] * 2 + P[3] * 2};
  float* outF = (float*)d_out;

  Bump al{(char*)d_ws, 0};
  unsigned short* Zpad = (unsigned short*)al.alloc(1024 * 2);   // conv OOB zero page
  unsigned short* X[5];
  float* W2[5];
  float* fu[5] = {nullptr, nullptr, nullptr, nullptr, nullptr};   // flow_up at level-l res
  float* cu[5] = {nullptr, nullptr, nullptr, nullptr, nullptr};   // ctx_up  at level-l res
  for (int l = 1; l <= 4; ++l) X[l] = (unsigned short*)al.alloc((size_t)P[l] * LCFG[l].Cx * 2);
  for (int l = 1; l <= 3; ++l) {
    W2[l] = (float*)al.alloc((size_t)P[l] * CHANS_L[l] * 4);
    fu[l] = (float*)al.alloc((size_t)P[l] * 2 * 4);
    cu[l] = (float*)al.alloc((size_t)P[l] * 32 * 4);
  }
  unsigned short* R0 = (unsigned short*)al.alloc((size_t)P[1] * 128 * 2);
  unsigned short* R1 = (unsigned short*)al.alloc((size_t)P[1] * 128 * 2);
  float* refbuf = (float*)al.alloc((size_t)P[1] * 2 * 4);
  float* acc = (float*)al.alloc((size_t)(4 * B + 2) * 4);  // acc1 | acc2 | stats
  float* acc1 = acc, *acc2 = acc + 2 * B, *stats = acc + 4 * B;

  zero_u16x8<<<g1(1024 / 8, 128), 128, 0, stream>>>(Zpad, 1024 / 8);

  // ---- coarse-to-fine decoder ----
  for (int lvl = 4; lvl >= 1; --lvl) {
    const LvlCfg& cfg = LCFG[lvl];
    const int H = Hl[lvl], W = Wl[lvl], C = CHANS_L[lvl];
    const long np = P[lvl];
    const float* f1 = (const float*)d_in[lvl];
    const float* f2 = (const float*)d_in[5 + lvl];

    zero_u16x8<<<g1(np * cfg.Cx / 8, 256), 256, 0, stream>>>(X[lvl], np * cfg.Cx / 8);

    const float* w2p;
    if (lvl < 4) {
      warp_kernel<<<g1(np, 128), 128, 0, stream>>>(f2, fu[lvl], W2[lvl], B, H, W, C);
      w2p = W2[lvl];
      f32_to_x<<<g1(np * 32, 256), 256, 0, stream>>>(cu[lvl], X[lvl], cfg.Cx, 0, np, 32);
      f32_to_x<<<g1(np * 2, 256), 256, 0, stream>>>(fu[lvl], X[lvl], cfg.Cx, 32, np, 2);
    } else {
      w2p = f2;
    }

    // normalization stats (shared scalar m, 1/std)
    long perImage = (long)H * W * C;
    zero_f32<<<g1(4 * B + 2, 64), 64, 0, stream>>>(acc, 4 * B + 2);
    dim3 rg((unsigned)min((long)1024, (perImage + 127) / 128), (unsigned)B);
    reduce_sums<<<rg, 128, 0, stream>>>(f1, perImage, acc1);
    reduce_sums<<<rg, 128, 0, stream>>>(w2p, perImage, acc2);
    finalize_stats<<<1, 1, 0, stream>>>(acc1, acc2, perImage, B, stats);

    cost_volume<<<g1(np * 81, 256), 256, 0, stream>>>(f1, w2p, stats, X[lvl],
                                                      cfg.Cx, cfg.cvOff, B, H, W, C);
    f32_to_x<<<g1(np * C, 256), 256, 0, stream>>>(f1, X[lvl], cfg.Cx, cfg.f1Off, np, C);

    // dense-concat conv stack
    for (int k = 0; k < 5; ++k) {
      int moff[10], roff[10], slen[10], ns = 0;
      if (lvl < 4) {
        moff[ns]=0;  roff[ns]=0;   slen[ns]=32; ns++;
        moff[ns]=32; roff[ns]=32;  slen[ns]=2;  ns++;
        moff[ns]=cfg.cvOff; roff[ns]=34;  slen[ns]=81; ns++;
        moff[ns]=cfg.f1Off; roff[ns]=115; slen[ns]=C;  ns++;
      } else {
        moff[ns]=0;         roff[ns]=0;  slen[ns]=81;  ns++;
        moff[ns]=cfg.f1Off; roff[ns]=81; slen[ns]=128; ns++;
      }
      int rb = CINREAL0[lvl];
      for (int j = 0; j < k; ++j) { moff[ns]=cfg.o[j]; roff[ns]=rb; slen[ns]=COUT[j]; ns++; rb += COUT[j]; }
      unsigned short* Wp = do_pack(stream, al, (const float*)d_in[flow_w(lvl, k)],
                                   cfg.o[k], COUT[k], rb, ns, moff, roff, slen);
      launch_conv(stream, X[lvl], cfg.Cx, Zpad, cfg.o[k], 1, Wp,
                  (const float*)d_in[flow_b(lvl, k)], COUT[k], B, H, W,
                  X[lvl] + cfg.o[k], cfg.Cx, nullptr, 0, nullptr, 1);
    }
    // flow head: context(32) -> 2, += flow_up
    {
      int moff = 0, roff = 0, slen = 32;
      unsigned short* Wp = do_pack(stream, al, (const float*)d_in[flow_w(lvl, 5)],
                                   32, 2, 32, 1, &moff, &roff, &slen);
      launch_conv(stream, X[lvl] + cfg.o[4], cfg.Cx, Zpad, 32, 1, Wp,
                  (const float*)d_in[flow_b(lvl, 5)], 2, B, H, W,
                  nullptr, 0, outF + outOff[lvl], 2,
                  (lvl < 4) ? fu[lvl] : nullptr, 0);
    }
    if (lvl > 1) {
      upsample2x_flow<<<g1(P[lvl - 1], 128), 128, 0, stream>>>(outF + outOff[lvl],
                                                               fu[lvl - 1], B, H, W);
      deconv4x4<<<g1(P[lvl - 1], 64), 64, 0, stream>>>(X[lvl], cfg.Cx, cfg.o[4],
          (const float*)d_in[ctx_w(lvl)], (const float*)d_in[ctx_b(lvl)],
          cu[lvl - 1], B, H, W);
    }
  }

  // ---- refinement (dilated convs on [context, flow] at level-1 res) ----
  {
    const long np = P[1];
    zero_u16x8<<<g1(np * 128 / 8, 256), 256, 0, stream>>>(R0, np * 128 / 8);
    zero_u16x8<<<g1(np * 128 / 8, 256), 256, 0, stream>>>(R1, np * 128 / 8);
    copy_bf_x<<<g1(np * 32, 256), 256, 0, stream>>>(X[1], LCFG[1].Cx, LCFG[1].o[4],
                                                    R0, 128, 0, np, 32);
    f32_to_x<<<g1(np * 2, 256), 256, 0, stream>>>(outF, R0, 128, 32, np, 2);

    const int RCOUT[6] = {128, 128, 128, 96, 64, 32};
    const int RDIL[6]  = {1, 2, 4, 8, 16, 1};
    unsigned short *rin = R0, *rout = R1;
    int rc = 64, rc_real = 34;
    for (int k = 0; k < 6; ++k) {
      int moff[10], roff[10], slen[10], ns = 0;
      if (k == 0) { moff[0]=0; roff[0]=0; slen[0]=32; moff[1]=32; roff[1]=32; slen[1]=2; ns=2; }
      else { moff[0]=0; roff[0]=0; slen[0]=rc; ns=1; }
      unsigned short* Wp = do_pack(stream, al, (const float*)d_in[ref_w(k)],
                                   rc, RCOUT[k], rc_real, ns, moff, roff, slen);
      launch_conv(stream, rin, 128, Zpad, rc, RDIL[k], Wp, (const float*)d_in[ref_b(k)],
                  RCOUT[k], B, Hl[1], Wl[1], rout, 128, nullptr, 0, nullptr, 1);
      unsigned short* t = rin; rin = rout; rout = t;
      rc = RCOUT[k]; rc_real = RCOUT[k];
    }
    {
      int moff = 0, roff = 0, slen = 32;
      unsigned short* Wp = do_pack(stream, al, (const float*)d_in[ref_w(6)],
                                   32, 2, 32, 1, &moff, &roff, &slen);
      launch_conv(stream, rin, 128, Zpad, 32, 1, Wp, (const float*)d_in[ref_b(6)],
                  2, B, Hl[1], Wl[1], nullptr, 0, refbuf, 2, nullptr, 0);
    }
    add_inplace<<<g1(np * 2, 256), 256, 0, stream>>>(outF, refbuf, np * 2);
  }
}
